// DifferentiableParticleFilter_82179904242178
// MI455X (gfx1250) — matrix operand order repaired and checked by reading, not compiled, not run on hardware
//
#include <hip/hip_runtime.h>
#include <hip/hip_bf16.h>
#include <math.h>

// ---------------------------------------------------------------------------
// Differentiable particle filter step, MI455X (gfx1250, wave32).
//  ka: fold h_t into biases (h is broadcast across particles), alpha/R/scales
//  kb: per-particle transition + skew-normal loglik -> lw[i], stateT (bf16, 64xN)
//  kc: global max(lw) + E_j = exp(2*(lw_j - max))           (softmax shift-inv.)
//  kd: out_partial = [e^{2lw_j} * (-ln(u+eps)+eps)^{-2}] @ state  via
//      v_wmma_f32_16x16x32_bf16, split-K x4, ones-column gives row sums
//  ke: sum split-K partials, divide by column 49, emit (N,49) fp32
// ---------------------------------------------------------------------------

typedef __bf16 v16bf __attribute__((ext_vector_type(16)));
typedef float  v8f   __attribute__((ext_vector_type(8)));

#define NPART 8192
#define KSPLIT 4
#define KPER  (NPART / KSPLIT)       // 2048
#define NCHUNK (KPER / 32)           // 64 wmma k-steps per split

// ws layout (float offsets)
#define OFF_PRE   0                  // [0..167] ch_d1(64) ch_rt(32) ch_g(32) ch_c(32) alpha Rsrc scales(5)
#define OFF_LW    256                // 8192
#define OFF_EV    8448               // 8192
#define OFF_ACC   16640              // 4 * 8192 * 64
#define OFF_STB   (16640 + 4*8192*64) // bf16 stateT 64 x 8192 starts here (as float idx)

__device__ __forceinline__ float siluf(float x)     { return x / (1.0f + __expf(-x)); }
__device__ __forceinline__ float softplusf(float x) { return (x > 20.0f) ? x : __logf(1.0f + __expf(x)); }
__device__ __forceinline__ float norm_logcdf(float x) {
  float t = -x * 0.70710678f;
  if (t < 8.0f) return __logf(0.5f * erfcf(t));
  return -0.5f * x * x - __logf(-x) - 0.91893853f;   // asymptotic tail
}

// ---------------------------------------------------------------- kernel A --
__global__ __launch_bounds__(64) void ka_pre(
    const float* __restrict__ h, const float* __restrict__ W_rt1, const float* __restrict__ b_rt1,
    const float* __restrict__ W_d1, const float* __restrict__ b_d1,
    const float* __restrict__ W_g,  const float* __restrict__ b_g,
    const float* __restrict__ W_c,  const float* __restrict__ b_c,
    const float* __restrict__ log_R, const float* __restrict__ log_obs_scale,
    const float* __restrict__ W_a1, const float* __restrict__ b_a1,
    const float* __restrict__ W_a2, const float* __restrict__ b_a2,
    float* __restrict__ pre)
{
  const int t = threadIdx.x;
  if (t < 64) {
    float a = b_d1[t];
    for (int k = 0; k < 64; ++k) a += W_d1[t*112 + k] * h[k];
    pre[t] = a;
  }
  if (t < 32) {
    float a = b_rt1[t], g = b_g[t], c = b_c[t];
    for (int k = 0; k < 64; ++k) {
      float hk = h[k];
      a += W_rt1[t*80  + k] * hk;
      g += W_g  [t*112 + k] * hk;
      c += W_c  [t*112 + k] * hk;
    }
    pre[64 + t] = a; pre[96 + t] = g; pre[128 + t] = c;
  }
  if (t == 0) {
    float alpha = b_a2[0];
    for (int m = 0; m < 16; ++m) {
      float a = b_a1[m];
      for (int k = 0; k < 64; ++k) a += W_a1[m*64 + k] * h[k];
      alpha += W_a2[m] * siluf(a);
    }
    pre[160] = alpha;
    pre[161] = fminf(2.5f, fmaxf(0.15f, __expf(log_R[0])));
  }
  if (t < 5) pre[162 + t] = softplusf(log_obs_scale[t]);
}

// ---------------------------------------------------------------- kernel B --
__global__ __launch_bounds__(256) void kb_particle(
    const float* __restrict__ rh, const float* __restrict__ rl,
    const float* __restrict__ z_, const float* __restrict__ logits_,
    const float* __restrict__ logw, const float* __restrict__ obs_p,
    const float* __restrict__ eps_h, const float* __restrict__ eps_l,
    const float* __restrict__ embed,
    const float* __restrict__ W_rt1, const float* __restrict__ W_rt2, const float* __restrict__ b_rt2,
    const float* __restrict__ W_d1,  const float* __restrict__ W_d2,  const float* __restrict__ b_d2,
    const float* __restrict__ W_d3,  const float* __restrict__ b_d3,
    const float* __restrict__ W_g,   const float* __restrict__ W_c,
    const float* __restrict__ pre,
    float* __restrict__ lw_out, __bf16* __restrict__ stateT)
{
  const int i = blockIdx.x * 256 + threadIdx.x;
  if (i >= NPART) return;

  // regime softmax (15) -> rp[:5] @ embed
  float lg[15];
  #pragma unroll
  for (int k = 0; k < 15; ++k) lg[k] = logits_[i*15 + k];
  float m = lg[0];
  #pragma unroll
  for (int k = 1; k < 15; ++k) m = fmaxf(m, lg[k]);
  float ex[15], s = 0.0f;
  #pragma unroll
  for (int k = 0; k < 15; ++k) { ex[k] = __expf(lg[k] - m); s += ex[k]; }
  const float inv = 1.0f / s;
  float remb[16];
  #pragma unroll
  for (int e = 0; e < 16; ++e) {
    float a = 0.0f;
    #pragma unroll
    for (int k = 0; k < 5; ++k) a += ex[k] * embed[k*16 + e];
    remb[e] = a * inv;
  }

  // regime transition MLP (h part pre-folded into pre[64..95])
  float srt[32];
  for (int j = 0; j < 32; ++j) {
    float a = pre[64 + j];
    #pragma unroll
    for (int e = 0; e < 16; ++e) a += W_rt1[j*80 + 64 + e] * remb[e];
    srt[j] = siluf(a);
  }
  float nlg[15];
  #pragma unroll
  for (int k = 0; k < 15; ++k) nlg[k] = lg[k];
  for (int k = 0; k < 5; ++k) {
    float a = b_rt2[k];
    for (int j = 0; j < 32; ++j) a += W_rt2[k*32 + j] * srt[j];
    nlg[k] = 0.7f * lg[k] + 0.3f * a;
  }

  // dynamics MLP
  float zv[32];
  #pragma unroll
  for (int q = 0; q < 32; ++q) zv[q] = z_[i*32 + q];
  float s1[64];
  for (int j = 0; j < 64; ++j) {
    float a = pre[j];
    #pragma unroll
    for (int e = 0; e < 16; ++e) a += W_d1[j*112 + 64 + e] * remb[e];
    #pragma unroll
    for (int q = 0; q < 32; ++q) a += W_d1[j*112 + 80 + q] * zv[q];
    s1[j] = siluf(a);
  }
  float s2[32];
  for (int k = 0; k < 32; ++k) {
    float a = b_d2[k];
    for (int j = 0; j < 64; ++j) a += W_d2[k*64 + j] * s1[j];
    s2[k] = siluf(a);
  }
  float dp[4];
  for (int k = 0; k < 4; ++k) {
    float a = b_d3[k];
    for (int j = 0; j < 32; ++j) a += W_d3[k*32 + j] * s2[j];
    dp[k] = a;
  }
  const float sigh = softplusf(dp[2]) + 0.01f;
  const float sigl = softplusf(dp[3]) + 0.01f;
  const float nh = fmaxf(0.0f, rh[i] + dp[0] + sigh * eps_h[i]);
  const float nl = fmaxf(0.0f, rl[i] + dp[1] + sigl * eps_l[i]);

  float nz[32];
  for (int j = 0; j < 32; ++j) {
    float ag = pre[96 + j], ac = pre[128 + j];
    #pragma unroll
    for (int e = 0; e < 16; ++e) {
      ag += W_g[j*112 + 64 + e] * remb[e];
      ac += W_c[j*112 + 64 + e] * remb[e];
    }
    #pragma unroll
    for (int q = 0; q < 32; ++q) {
      ag += W_g[j*112 + 80 + q] * zv[q];
      ac += W_c[j*112 + 80 + q] * zv[q];
    }
    const float gate = 1.0f / (1.0f + __expf(-ag));
    nz[j] = gate * zv[j] + (1.0f - gate) * tanhf(ac);
  }

  // observation loglik
  float m2 = nlg[0];
  #pragma unroll
  for (int k = 1; k < 15; ++k) m2 = fmaxf(m2, nlg[k]);
  float sum2 = 0.0f, e5[5];
  #pragma unroll
  for (int k = 0; k < 15; ++k) {
    float e = __expf(nlg[k] - m2);
    sum2 += e;
    if (k < 5) e5[k] = e;
  }
  const float alpha = pre[160], Rsrc = pre[161];
  float dsc = 0.0f;
  #pragma unroll
  for (int k = 0; k < 5; ++k) dsc += e5[k] * pre[162 + k];
  dsc /= sum2;
  const float R = fminf(4.0f, fmaxf(0.15f, Rsrc * dsc));
  const float zz = (obs_p[0] - nh) / R;
  const float ll = 0.69314718f - __logf(R) - 0.91893853f - 0.5f * zz * zz
                 + norm_logcdf(alpha * zz);
  lw_out[i] = logw[i] + ll;

  // transposed bf16 state, padded to 64 cols, col 49 == 1.0 (row-sum column)
  stateT[(size_t)0 * NPART + i] = (__bf16)nh;
  stateT[(size_t)1 * NPART + i] = (__bf16)nl;
  #pragma unroll
  for (int j = 0; j < 32; ++j) stateT[(size_t)(2 + j) * NPART + i] = (__bf16)nz[j];
  #pragma unroll
  for (int k = 0; k < 15; ++k) stateT[(size_t)(34 + k) * NPART + i] = (__bf16)nlg[k];
  stateT[(size_t)49 * NPART + i] = (__bf16)1.0f;
  #pragma unroll
  for (int c = 50; c < 64; ++c) stateT[(size_t)c * NPART + i] = (__bf16)0.0f;
}

// ---------------------------------------------------------------- kernel C --
__global__ __launch_bounds__(1024) void kc_norm(const float* __restrict__ lw,
                                                float* __restrict__ Ev)
{
  __shared__ float red[1024];
  const int t = threadIdx.x;
  float m = -1e30f;
  #pragma unroll
  for (int k = 0; k < 8; ++k) m = fmaxf(m, lw[t + k*1024]);
  red[t] = m;
  __syncthreads();
  for (int s = 512; s > 0; s >>= 1) {
    if (t < s) red[t] = fmaxf(red[t], red[t + s]);
    __syncthreads();
  }
  const float lwmax = red[0];
  #pragma unroll
  for (int k = 0; k < 8; ++k) {
    const int j = t + k*1024;
    Ev[j] = __expf(2.0f * (lw[j] - lwmax));   // 1/TAU == 2
  }
}

// ---------------------------------------------------------------- kernel D --
// grid (128, 4), block 128 (4 waves). Wave w of block b owns row tile b*4+w
// (16 rows) and K-range blockIdx.y*2048. Single pass over u_gumbel.
__global__ __launch_bounds__(128) void kd_softres(
    const float* __restrict__ u, const float* __restrict__ Ev,
    const __bf16* __restrict__ stateT, float* __restrict__ acc)
{
  __shared__ float Elds[NPART];
  const int t = threadIdx.x;
  for (int k = 0; k < NPART / (128 * 4); ++k) {      // 16 iters of float4
    const int idx = (k * 128 + t) * 4;
    *(float4*)&Elds[idx] = *(const float4*)&Ev[idx];
  }
  __syncthreads();

  const int lane    = t & 31;
  const int wave    = t >> 5;
  const int tile    = blockIdx.x * 4 + wave;         // 0..511
  const int rowBase = tile * 16;
  const int kc0     = blockIdx.y * KPER;
  const int row     = rowBase + (lane & 15);
  const int kb      = (lane < 16) ? 0 : 16;          // A-frag K base (contiguous, matches B)
  const int koff    = (lane < 16) ? 0 : 16;          // B-frag K base
  const int colLane = lane & 15;
  const float* uRow = u + (size_t)row * NPART + kc0 + kb;

  v8f a0 = {}, a1 = {}, a2 = {}, a3 = {};

  for (int c = 0; c < NCHUNK; ++c) {
    const int kc = kc0 + c * 32;
    // 16 u values per lane in A-fragment order: K = kb+{0..15}
    const float4 u0 = *(const float4*)(uRow + c*32);
    const float4 u1 = *(const float4*)(uRow + c*32 + 4);
    const float4 u2 = *(const float4*)(uRow + c*32 + 8);
    const float4 u3 = *(const float4*)(uRow + c*32 + 12);
    const float4 e0 = *(const float4*)&Elds[kc + kb];
    const float4 e1 = *(const float4*)&Elds[kc + kb + 4];
    const float4 e2 = *(const float4*)&Elds[kc + kb + 8];
    const float4 e3 = *(const float4*)&Elds[kc + kb + 12];

    float uu[16] = {u0.x,u0.y,u0.z,u0.w, u1.x,u1.y,u1.z,u1.w,
                    u2.x,u2.y,u2.z,u2.w, u3.x,u3.y,u3.z,u3.w};
    float ee[16] = {e0.x,e0.y,e0.z,e0.w, e1.x,e1.y,e1.z,e1.w,
                    e2.x,e2.y,e2.z,e2.w, e3.x,e3.y,e3.z,e3.w};
    v16bf A;
    #pragma unroll
    for (int q = 0; q < 16; ++q) {
      // w = e^{2 lw_j} * (-ln(u+1e-10)+1e-10)^-2  == exp((lw+g)/tau), tau=0.5
      const float llu = 1e-10f - __logf(uu[q] + 1e-10f);
      const float w   = ee[q] * __builtin_amdgcn_rcpf(llu * llu);
      A[q] = (__bf16)w;
    }

    // B fragments: 16 contiguous bf16 per lane from transposed state
    const __bf16* sB = stateT + (size_t)(kc + koff);
    const v16bf B0 = *(const v16bf*)(sB + (size_t)(colLane)      * NPART);
    const v16bf B1 = *(const v16bf*)(sB + (size_t)(16 + colLane) * NPART);
    const v16bf B2 = *(const v16bf*)(sB + (size_t)(32 + colLane) * NPART);
    const v16bf B3 = *(const v16bf*)(sB + (size_t)(48 + colLane) * NPART);

    a0 = __builtin_amdgcn_wmma_f32_16x16x32_bf16(false, A, false, B0, (short)0, a0, false, false);
    a1 = __builtin_amdgcn_wmma_f32_16x16x32_bf16(false, A, false, B1, (short)0, a1, false, false);
    a2 = __builtin_amdgcn_wmma_f32_16x16x32_bf16(false, A, false, B2, (short)0, a2, false, false);
    a3 = __builtin_amdgcn_wmma_f32_16x16x32_bf16(false, A, false, B3, (short)0, a3, false, false);
  }

  // store partial accumulator tile (C layout: VGPR r -> M=r (lanes<16), M=8+r)
  float* ab = acc + ((size_t)blockIdx.y * NPART + rowBase) * 64;
  #pragma unroll
  for (int r = 0; r < 8; ++r) {
    const int rr = r + ((lane < 16) ? 0 : 8);
    ab[rr*64 +      colLane] = a0[r];
    ab[rr*64 + 16 + colLane] = a1[r];
    ab[rr*64 + 32 + colLane] = a2[r];
    ab[rr*64 + 48 + colLane] = a3[r];
  }
}

// ---------------------------------------------------------------- kernel E --
__global__ __launch_bounds__(64) void ke_final(const float* __restrict__ acc,
                                               float* __restrict__ out)
{
  const int i = blockIdx.x;
  const int c = threadIdx.x;
  const size_t S = (size_t)NPART * 64;
  const size_t base = (size_t)i * 64;
  const float v = acc[base + c] + acc[S + base + c] + acc[2*S + base + c] + acc[3*S + base + c];
  const float d = acc[base + 49] + acc[S + base + 49] + acc[2*S + base + 49] + acc[3*S + base + 49];
  if (c < 49) out[(size_t)i * 49 + c] = v / d;
}

// --------------------------------------------------------------------------
extern "C" void kernel_launch(void* const* d_in, const int* in_sizes, int n_in,
                              void* d_out, int out_size, void* d_ws, size_t ws_size,
                              hipStream_t stream) {
  const float* rh      = (const float*)d_in[0];
  const float* rl      = (const float*)d_in[1];
  const float* z_      = (const float*)d_in[2];
  const float* logits  = (const float*)d_in[3];
  const float* logw    = (const float*)d_in[4];
  const float* h_t     = (const float*)d_in[5];
  const float* obs     = (const float*)d_in[6];
  const float* eps_h   = (const float*)d_in[7];
  const float* eps_l   = (const float*)d_in[8];
  const float* u_gum   = (const float*)d_in[9];
  const float* embed   = (const float*)d_in[10];
  const float* W_rt1   = (const float*)d_in[11];
  const float* b_rt1   = (const float*)d_in[12];
  const float* W_rt2   = (const float*)d_in[13];
  const float* b_rt2   = (const float*)d_in[14];
  const float* W_d1    = (const float*)d_in[15];
  const float* b_d1    = (const float*)d_in[16];
  const float* W_d2    = (const float*)d_in[17];
  const float* b_d2    = (const float*)d_in[18];
  const float* W_d3    = (const float*)d_in[19];
  const float* b_d3    = (const float*)d_in[20];
  const float* W_g     = (const float*)d_in[21];
  const float* b_g     = (const float*)d_in[22];
  const float* W_c     = (const float*)d_in[23];
  const float* b_c     = (const float*)d_in[24];
  const float* log_R   = (const float*)d_in[25];
  const float* log_osc = (const float*)d_in[26];
  const float* W_a1    = (const float*)d_in[27];
  const float* b_a1    = (const float*)d_in[28];
  const float* W_a2    = (const float*)d_in[29];
  const float* b_a2    = (const float*)d_in[30];

  float*  ws     = (float*)d_ws;
  float*  pre    = ws + OFF_PRE;
  float*  lw     = ws + OFF_LW;
  float*  Ev     = ws + OFF_EV;
  float*  acc    = ws + OFF_ACC;
  __bf16* stateT = (__bf16*)(ws + OFF_STB);

  ka_pre<<<1, 64, 0, stream>>>(h_t, W_rt1, b_rt1, W_d1, b_d1, W_g, b_g, W_c, b_c,
                               log_R, log_osc, W_a1, b_a1, W_a2, b_a2, pre);
  kb_particle<<<NPART / 256, 256, 0, stream>>>(
      rh, rl, z_, logits, logw, obs, eps_h, eps_l, embed,
      W_rt1, W_rt2, b_rt2, W_d1, W_d2, b_d2, W_d3, b_d3, W_g, W_c,
      pre, lw, stateT);
  kc_norm<<<1, 1024, 0, stream>>>(lw, Ev);
  kd_softres<<<dim3(128, KSPLIT), 128, 0, stream>>>(u_gum, Ev, stateT, acc);
  ke_final<<<NPART, 64, 0, stream>>>(acc, (float*)d_out);
}
